// ScaledDotProductAttention_31052613550604
// MI455X (gfx1250) — matrix operand order repaired and checked by reading, not compiled, not run on hardware
//
#include <hip/hip_runtime.h>
#include <hip/hip_bf16.h>

// ScaledDotProductAttention for MI455X (gfx1250), wave32 + WMMA bf16 path.
//
// ws layout (element offsets, all bf16 stored as u16):
//   Qbf  [4096][64]   (pre-scaled by 1/sqrt(64))      512 KB
//   Kbf  [4096][64]                                   512 KB
//   Vt   [64][4096]   (transposed V)                  512 KB
//   Xbf  [4096][512]  (bf16 embeddings)               4 MB
//   Wbf  [3][64][512] (bf16 Wq,Wk,Wv)                 192 KB

#define SEQ    4096
#define DMODEL 512
#define DK     64

typedef __bf16 v16bf __attribute__((ext_vector_type(16)));
typedef float  v8f   __attribute__((ext_vector_type(8)));

union FragBF {
  v16bf bf;
  unsigned int u[8];
};

__device__ __forceinline__ unsigned short f2bf(float f) {
  // scalar fp32 -> bf16 RNE (used only in small epilogues)
  unsigned int u = __float_as_uint(f);
  u = (u + 0x7FFFu + ((u >> 16) & 1u)) >> 16;
  return (unsigned short)u;
}

#if defined(__has_builtin) && __has_builtin(__builtin_amdgcn_cvt_pk_bf16_f32)
// one v_cvt_pk_bf16_f32 per pair (RNE in hardware)
__device__ __forceinline__ unsigned int packbf(float lo, float hi) {
  typedef __bf16 v2bf __attribute__((ext_vector_type(2)));
  union { v2bf v; unsigned int u; } cvt;
  cvt.v = __builtin_amdgcn_cvt_pk_bf16_f32(lo, hi);
  return cvt.u;
}
#else
__device__ __forceinline__ unsigned int packbf(float lo, float hi) {
  return (unsigned int)f2bf(lo) | ((unsigned int)f2bf(hi) << 16);
}
#endif

// lane-XOR shuffle within the wave via ds_swizzle (group-of-32 encoding:
// offset = {0, xor[14:10], or[9:5]=0, and[4:0]=0x1f})
template<int MASK>
__device__ __forceinline__ float swzx(float x) {
  int i = __builtin_amdgcn_ds_swizzle(__float_as_int(x), (MASK << 10) | 0x1f);
  return __int_as_float(i);
}
// reductions across the 16 lanes of each half-wave (XOR 1,2,4,8 never crosses
// the lane-16 boundary, matching the C-matrix row layout).
__device__ __forceinline__ float rowmax16(float v) {
  v = fmaxf(v, swzx<1>(v));
  v = fmaxf(v, swzx<2>(v));
  v = fmaxf(v, swzx<4>(v));
  v = fmaxf(v, swzx<8>(v));
  return v;
}
__device__ __forceinline__ float rowsum16(float v) {
  v += swzx<1>(v);
  v += swzx<2>(v);
  v += swzx<4>(v);
  v += swzx<8>(v);
  return v;
}

// K-offset of the bf16 pair held in A-fragment VGPR v (contiguous order, matching B):
// v0..7 -> K = 2v (+16 for lanes 16..31)
__device__ __forceinline__ int a_k0(int v, int half) {
  return 2 * v + 16 * half;
}

// ---------------------------------------------------------------------------
// Kernel 0: bulk fp32 -> bf16 conversion (one float4 -> two packed u32 / thread)
// ---------------------------------------------------------------------------
__global__ void __launch_bounds__(256)
cvt_bf16_kernel(const float* __restrict__ src, unsigned int* __restrict__ dst,
                int n4) {
  const int i = blockIdx.x * blockDim.x + threadIdx.x;
  if (i < n4) {
    const float4 x = reinterpret_cast<const float4*>(src)[i];
    dst[2 * i]     = packbf(x.x, x.y);
    dst[2 * i + 1] = packbf(x.z, x.w);
  }
}

// ---------------------------------------------------------------------------
// Kernel 1: Q/K/V projections (pure bf16 fragment loads, zero conversions in
// the hot loop).  One wave computes a 16-row tile of all three outputs:
// 12 persistent accumulators (3 weights x 4 N-tiles), K looped by 32.
// ---------------------------------------------------------------------------
__global__ void __launch_bounds__(128)
proj_qkv_kernel(const unsigned short* __restrict__ Xbf,
                const unsigned short* __restrict__ Wbf,
                const float* __restrict__ bq,
                const float* __restrict__ bk,
                const float* __restrict__ bv,
                unsigned short* __restrict__ Qbf,
                unsigned short* __restrict__ Kbf,
                unsigned short* __restrict__ Vt) {
  const int lane = threadIdx.x & 31;
  const int wave = (blockIdx.x * blockDim.x + threadIdx.x) >> 5;
  const int m    = lane & 15;
  const int half = lane >> 4;
  const int row0 = wave * 16;

  const float* Bs[3] = {bq, bk, bv};

  v8f acc[12];
#pragma unroll
  for (int i = 0; i < 12; ++i)
#pragma unroll
    for (int r = 0; r < 8; ++r) acc[i][r] = 0.0f;

  const unsigned short* xrow = Xbf + (size_t)(row0 + m) * DMODEL;

  for (int kb = 0; kb < DMODEL; kb += 32) {
    FragBF a;
#pragma unroll
    for (int v = 0; v < 8; ++v)
      a.u[v] = *reinterpret_cast<const unsigned int*>(xrow + kb + a_k0(v, half));

#pragma unroll
    for (int w = 0; w < 3; ++w) {
#pragma unroll
      for (int nt = 0; nt < 4; ++nt) {
        // B fragment: B[k][n] = W[n][k]; lane n = lane&15, K pairs contiguous
        const unsigned short* wrow =
            Wbf + (size_t)(w * 64 + nt * 16 + m) * DMODEL;
        FragBF b;
#pragma unroll
        for (int v = 0; v < 8; ++v)
          b.u[v] = *reinterpret_cast<const unsigned int*>(wrow + kb + 2 * v +
                                                          16 * half);
        acc[w * 4 + nt] = __builtin_amdgcn_wmma_f32_16x16x32_bf16(
            false, a.bf, false, b.bf, (short)0, acc[w * 4 + nt], false, false);
      }
    }
  }

  // epilogue: +bias, Q scaled by 1/sqrt(64), stored bf16 (V transposed)
#pragma unroll
  for (int w = 0; w < 3; ++w) {
#pragma unroll
    for (int nt = 0; nt < 4; ++nt) {
      const int n = nt * 16 + m;
      const float bias = Bs[w][n];
#pragma unroll
      for (int r = 0; r < 8; ++r) {
        const int row = row0 + r + 8 * half;
        const float val = acc[w * 4 + nt][r] + bias;
        if (w == 0) {
          Qbf[(size_t)row * DK + n] = f2bf(val * 0.125f);
        } else if (w == 1) {
          Kbf[(size_t)row * DK + n] = f2bf(val);
        } else {
          Vt[(size_t)n * SEQ + row] = f2bf(val);
        }
      }
    }
  }
}

// ---------------------------------------------------------------------------
// Kernel 2: causal flash attention.  One wave owns a 16-row query tile and
// streams 32-key blocks up to the diagonal with online softmax.
// ---------------------------------------------------------------------------
__global__ void __launch_bounds__(128)
flash_attn_kernel(const unsigned short* __restrict__ Qbf,
                  const unsigned short* __restrict__ Kbf,
                  const unsigned short* __restrict__ Vt,
                  float* __restrict__ Out) {
  __shared__ __attribute__((aligned(16))) unsigned short Plds[4][16][32];

  const int lane  = threadIdx.x & 31;
  const int wslot = threadIdx.x >> 5;
  const int qtile = (blockIdx.x * blockDim.x + threadIdx.x) >> 5;
  const int m     = lane & 15;
  const int half  = lane >> 4;
  const int qbase = qtile * 16;

  // Q A-fragments (dk = 64 -> two 16x32 fragments), loaded once
  FragBF aq[2];
  const unsigned short* qrow = Qbf + (size_t)(qbase + m) * DK;
#pragma unroll
  for (int f = 0; f < 2; ++f)
#pragma unroll
    for (int v = 0; v < 8; ++v)
      aq[f].u[v] =
          *reinterpret_cast<const unsigned int*>(qrow + f * 32 + a_k0(v, half));

  v8f o[4];
#pragma unroll
  for (int nt = 0; nt < 4; ++nt)
#pragma unroll
    for (int r = 0; r < 8; ++r) o[nt][r] = 0.0f;

  float Mr[8], Lr[8];
#pragma unroll
  for (int r = 0; r < 8; ++r) { Mr[r] = -3.0e38f; Lr[r] = 0.0f; }

  for (int kb = 0; kb < qbase + 16; kb += 32) {
    // ---- S = Q * K^T for two 16-key tiles (4 WMMAs) ----
    v8f s[2];
#pragma unroll
    for (int t = 0; t < 2; ++t) {
      const unsigned short* krow = Kbf + (size_t)(kb + t * 16 + m) * DK;
      FragBF b0, b1;
#pragma unroll
      for (int v = 0; v < 8; ++v) {
        const int k = 2 * v + 16 * half;
        b0.u[v] = *reinterpret_cast<const unsigned int*>(krow + k);
        b1.u[v] = *reinterpret_cast<const unsigned int*>(krow + 32 + k);
      }
      v8f z;
#pragma unroll
      for (int r = 0; r < 8; ++r) z[r] = 0.0f;
      z = __builtin_amdgcn_wmma_f32_16x16x32_bf16(false, aq[0].bf, false, b0.bf,
                                                  (short)0, z, false, false);
      s[t] = __builtin_amdgcn_wmma_f32_16x16x32_bf16(false, aq[1].bf, false, b1.bf,
                                                     (short)0, z, false, false);
    }

    // ---- causal mask (only blocks touching the diagonal) ----
    if (kb + 31 > qbase) {
#pragma unroll
      for (int t = 0; t < 2; ++t) {
        const int col = kb + t * 16 + m;
#pragma unroll
        for (int r = 0; r < 8; ++r) {
          const int row = qbase + r + 8 * half;
          if (col > row) s[t][r] = -3.0e38f;
        }
      }
    }

    // ---- online softmax; C-layout row = r + 8*half, lane = column ----
#pragma unroll
    for (int r = 0; r < 8; ++r) {
      const float bm   = rowmax16(fmaxf(s[0][r], s[1][r]));
      const float mnew = fmaxf(Mr[r], bm);
      const float corr = __expf(Mr[r] - mnew);
      Mr[r] = mnew;
      const float p0 = __expf(s[0][r] - mnew);
      const float p1 = __expf(s[1][r] - mnew);
      Lr[r] = Lr[r] * corr + rowsum16(p0 + p1);
#pragma unroll
      for (int nt = 0; nt < 4; ++nt) o[nt][r] *= corr;
      // park P (bf16) in this wave's LDS tile for C->A re-layout
      const unsigned int pk = packbf(p0, p1);
      Plds[wslot][r + 8 * half][m]      = (unsigned short)(pk & 0xFFFFu);
      Plds[wslot][r + 8 * half][16 + m] = (unsigned short)(pk >> 16);
    }

    // ---- read P back as a 16x32 bf16 A fragment ----
    FragBF ap;
#pragma unroll
    for (int v = 0; v < 8; ++v)
      ap.u[v] = *reinterpret_cast<const unsigned int*>(
          &Plds[wslot][m][a_k0(v, half)]);

    // ---- O += P * V (4 WMMAs; Vt gives contiguous K pairs) ----
#pragma unroll
    for (int nt = 0; nt < 4; ++nt) {
      const unsigned short* vrow = Vt + (size_t)(nt * 16 + m) * SEQ + kb;
      FragBF bvf;
#pragma unroll
      for (int v = 0; v < 8; ++v)
        bvf.u[v] =
            *reinterpret_cast<const unsigned int*>(vrow + 2 * v + 16 * half);
      o[nt] = __builtin_amdgcn_wmma_f32_16x16x32_bf16(
          false, ap.bf, false, bvf.bf, (short)0, o[nt], false, false);
    }
  }

  // ---- epilogue: O / L, fp32 out ----
#pragma unroll
  for (int r = 0; r < 8; ++r) {
    const float inv = 1.0f / Lr[r];
    const int row = qbase + r + 8 * half;
#pragma unroll
    for (int nt = 0; nt < 4; ++nt)
      Out[(size_t)row * DK + nt * 16 + m] = o[nt][r] * inv;
  }
}

extern "C" void kernel_launch(void* const* d_in, const int* in_sizes, int n_in,
                              void* d_out, int out_size, void* d_ws, size_t ws_size,
                              hipStream_t stream) {
  (void)in_sizes; (void)n_in; (void)out_size; (void)ws_size;

  const float* X  = (const float*)d_in[0];
  const float* Wq = (const float*)d_in[1];
  const float* bq = (const float*)d_in[2];
  const float* Wk = (const float*)d_in[3];
  const float* bk = (const float*)d_in[4];
  const float* Wv = (const float*)d_in[5];
  const float* bv = (const float*)d_in[6];

  unsigned short* Qbf = (unsigned short*)d_ws;
  unsigned short* Kbf = Qbf + (size_t)SEQ * DK;
  unsigned short* Vt  = Kbf + (size_t)SEQ * DK;
  unsigned short* Xbf = Vt + (size_t)DK * SEQ;
  unsigned short* Wbf = Xbf + (size_t)SEQ * DMODEL;   // [3][64][512]
  float* Out = (float*)d_out;

  // ---- bulk fp32 -> bf16 conversions (once, not per consuming wave) ----
  {
    const int nx4 = SEQ * DMODEL / 4;   // 524288 float4s
    cvt_bf16_kernel<<<dim3((nx4 + 255) / 256), dim3(256), 0, stream>>>(
        X, (unsigned int*)Xbf, nx4);
    const int nw4 = DK * DMODEL / 4;    // 8192 float4s per weight matrix
    cvt_bf16_kernel<<<dim3((nw4 + 255) / 256), dim3(256), 0, stream>>>(
        Wq, (unsigned int*)(Wbf + 0 * DK * DMODEL), nw4);
    cvt_bf16_kernel<<<dim3((nw4 + 255) / 256), dim3(256), 0, stream>>>(
        Wk, (unsigned int*)(Wbf + 1 * DK * DMODEL), nw4);
    cvt_bf16_kernel<<<dim3((nw4 + 255) / 256), dim3(256), 0, stream>>>(
        Wv, (unsigned int*)(Wbf + 2 * DK * DMODEL), nw4);
  }

  const dim3 blk(128);                 // 4 waves / block (wave32)
  const dim3 grd(SEQ / 16 / 4);        // one wave per 16-row tile

  proj_qkv_kernel<<<grd, blk, 0, stream>>>(Xbf, Wbf, bq, bk, bv,
                                           Qbf, Kbf, Vt);
  flash_attn_kernel<<<grd, blk, 0, stream>>>(Qbf, Kbf, Vt, Out);
}
